// SS2D_24361054503035
// MI455X (gfx1250) — compile-verified, matrix operand order checked by reading, not run on hardware
//
#include <hip/hip_runtime.h>

typedef __attribute__((ext_vector_type(16))) __bf16 v16bf;
typedef __attribute__((ext_vector_type(8)))  float  v8f;
typedef __attribute__((ext_vector_type(4)))  unsigned int v4u;
typedef __attribute__((ext_vector_type(8)))  int    v8i;
typedef __attribute__((ext_vector_type(4)))  int    v4i;

union FragU { uint4 q[2]; v16bf v; };

__device__ __forceinline__ unsigned short f2bf(float f) {
  union { float f; unsigned int u; } a; a.f = f;
  unsigned int u = a.u;
  u += 0x7FFFu + ((u >> 16) & 1u);   // round-to-nearest-even
  return (unsigned short)(u >> 16);
}

__device__ __forceinline__ float siluf(float x) {
  return x / (1.0f + __expf(-x));
}

#if defined(__gfx1250__) && __has_builtin(__builtin_amdgcn_tensor_load_to_lds)
#define USE_TDM 1
#else
#define USE_TDM 0
#endif

#if USE_TDM
// ---------------------------------------------------------------------------
// TDM 2D tile load: global [tileRows][32] bf16 (row stride K elements) -> LDS.
// pad_enable with interval=16 DWORDs (one 32-elem row) and amount=4 DWORDs
// (8 elems) produces a 40-element LDS row stride == the layout the WMMA
// fragment loads expect (conflict-free, 16B aligned).
// ---------------------------------------------------------------------------
__device__ __forceinline__ void tdm_load_2d(const unsigned short* gptr,
                                            unsigned ldsOff, int K, int tileRows)
{
  unsigned long long ga = (unsigned long long)(size_t)gptr;
  v4u g0;
  g0.x = 1u;                                        // count=1 (valid descriptor)
  g0.y = ldsOff;                                    // lds_addr (bytes)
  g0.z = (unsigned)(ga & 0xFFFFFFFFu);              // global_addr[31:0]
  g0.w = (unsigned)((ga >> 32) & 0x01FFFFFFu) | (2u << 30);  // addr[56:32] | type=2
  v8i g1 = {};
  g1[0] = (int)((1u << 16)        // data_size = 2 bytes
              | (1u << 20)        // pad_enable
              | (3u << 22)        // pad_interval: 16 DWORDs
              | (3u << 25));      // pad_amount:   4 DWORDs
  g1[1] = (int)((unsigned)K << 16);          // tensor_dim0[15:0] @ bits 63:48
  g1[2] = (int)((unsigned)tileRows << 16);   // tensor_dim1[15:0] @ bits 95:80
  g1[3] = (int)(32u << 16);                  // tile_dim0 = 32    @ bits 127:112
  g1[4] = tileRows;                          // tile_dim1         @ bits 143:128
  g1[5] = K;                                 // tensor_dim0_stride[31:0]
  g1[6] = 0;
  g1[7] = 0;
  v4i z4 = {};
#if __clang_major__ >= 23
  v8i z8 = {};
  __builtin_amdgcn_tensor_load_to_lds(g0, g1, z4, z4, z8, 0);
#else
  __builtin_amdgcn_tensor_load_to_lds(g0, g1, z4, z4, 0);
#endif
}
#endif

// ---------------------------------------------------------------------------
// Generic bf16 WMMA GEMM:  C[M][ldc] (f32) = A[M][K] (bf16) * Bw[n][k]^T
// Bw stored row-major [Npad][K] (== jax weight layout), so no transpose.
// Block tile 64(M) x 128(N), K-chunk 32, 8 waves, each wave 32x32 (4 WMMAs).
// Tiles staged into double-buffered LDS by the Tensor Data Mover; chunk i+1's
// DMA overlaps chunk i's WMMAs (2 TDM ops/chunk, in-order => wait tensorcnt<=2).
// Grid: (M/64, Npad/128). M % 64 == 0, K % 32 == 0 guaranteed by caller.
// ---------------------------------------------------------------------------
__global__ __launch_bounds__(256) void gemm_bf16_wmma(
    const unsigned short* __restrict__ A,
    const unsigned short* __restrict__ Bw,
    float* __restrict__ C, int K, int ldc, int Nvalid)
{
  __shared__ unsigned short lsA[2][64 * 40];   // padded stride 40 (80B = 20 banks)
  __shared__ unsigned short lsB[2][128 * 40];

  const int tid  = threadIdx.x;
  const int lane = tid & 31;
  const int wid  = tid >> 5;
  const int wm   = wid >> 2;      // 0..1  (M direction)
  const int wn   = wid & 3;       // 0..3  (N direction)
  const int hi   = lane >> 4;     // lane half
  const int lo   = lane & 15;

  const long m0 = (long)blockIdx.x * 64;
  const long n0 = (long)blockIdx.y * 128;

  v8f vz = {};
  v8f acc00 = vz, acc01 = vz, acc10 = vz, acc11 = vz;

  const int nk = K >> 5;

#if USE_TDM
  unsigned ldsA_off[2], ldsB_off[2];
  ldsA_off[0] = (unsigned)(size_t)(__attribute__((address_space(3))) unsigned short*)lsA[0];
  ldsA_off[1] = (unsigned)(size_t)(__attribute__((address_space(3))) unsigned short*)lsA[1];
  ldsB_off[0] = (unsigned)(size_t)(__attribute__((address_space(3))) unsigned short*)lsB[0];
  ldsB_off[1] = (unsigned)(size_t)(__attribute__((address_space(3))) unsigned short*)lsB[1];
  const unsigned short* agp = A  + m0 * (long)K;
  const unsigned short* bgp = Bw + n0 * (long)K;
  if (wid == 0) {                       // prologue: chunk 0 -> buffer 0
    tdm_load_2d(agp, ldsA_off[0], K, 64);
    tdm_load_2d(bgp, ldsB_off[0], K, 128);
  }
#else
  const int arow = tid >> 2;
  const int acs  = (tid & 3) * 8;
  const unsigned short* agp  = A  + (m0 + arow) * (long)K + acs;
  const unsigned short* bgp0 = Bw + (n0 + arow) * (long)K + acs;
  const unsigned short* bgp1 = Bw + (n0 + 64 + arow) * (long)K + acs;
#endif

  int buf = 0;
  for (int i = 0; i < nk; ++i) {
    const int k0 = i << 5;
#if USE_TDM
    if (i + 1 < nk) {
      if (wid == 0) {                   // issue chunk i+1 into the other buffer
        tdm_load_2d(agp + k0 + 32, ldsA_off[buf ^ 1], K, 64);
        tdm_load_2d(bgp + k0 + 32, ldsB_off[buf ^ 1], K, 128);
      }
      __builtin_amdgcn_s_wait_tensorcnt(2);   // chunk i done (in-order TDM)
    } else {
      __builtin_amdgcn_s_wait_tensorcnt(0);
    }
    __syncthreads();                    // publish chunk i's LDS to all waves
#else
    __syncthreads();
    uint4 va  = *(const uint4*)(agp  + k0);
    uint4 vb0 = *(const uint4*)(bgp0 + k0);
    uint4 vb1 = *(const uint4*)(bgp1 + k0);
    __builtin_prefetch(agp + k0 + 32, 0, 0);
    *(uint4*)&lsA[buf][arow * 40 + acs]        = va;
    *(uint4*)&lsB[buf][arow * 40 + acs]        = vb0;
    *(uint4*)&lsB[buf][(arow + 64) * 40 + acs] = vb1;
    __syncthreads();
#endif

    // A fragments: lane lo = M row, elements [0..7]=K(hi*8..+7), [8..15]=K(16+hi*8..+7)
    FragU a0, a1, b0, b1;
    {
      const unsigned short* p = &lsA[buf][(wm * 32 + lo) * 40 + hi * 8];
      a0.q[0] = *(const uint4*)p;        a0.q[1] = *(const uint4*)(p + 16);
      p += 16 * 40;
      a1.q[0] = *(const uint4*)p;        a1.q[1] = *(const uint4*)(p + 16);
    }
    {
      const unsigned short* p = &lsB[buf][(wn * 32 + lo) * 40 + hi * 8];
      b0.q[0] = *(const uint4*)p;        b0.q[1] = *(const uint4*)(p + 16);
      p += 16 * 40;
      b1.q[0] = *(const uint4*)p;        b1.q[1] = *(const uint4*)(p + 16);
    }

    acc00 = __builtin_amdgcn_wmma_f32_16x16x32_bf16(false, a0.v, false, b0.v, (short)0, acc00, false, false);
    acc01 = __builtin_amdgcn_wmma_f32_16x16x32_bf16(false, a0.v, false, b1.v, (short)0, acc01, false, false);
    acc10 = __builtin_amdgcn_wmma_f32_16x16x32_bf16(false, a1.v, false, b0.v, (short)0, acc10, false, false);
    acc11 = __builtin_amdgcn_wmma_f32_16x16x32_bf16(false, a1.v, false, b1.v, (short)0, acc11, false, false);

#if USE_TDM
    __syncthreads();                    // all waves done with buf before chunk i+2 overwrites it
#endif
    buf ^= 1;
  }

  // Store: lane lo = N col, VGPR r -> M = r + hi*8 within the 16x16 tile
  v8f accs[2][2] = {{acc00, acc01}, {acc10, acc11}};
  #pragma unroll
  for (int mt = 0; mt < 2; ++mt) {
    #pragma unroll
    for (int nt = 0; nt < 2; ++nt) {
      long n = n0 + wn * 32 + nt * 16 + lo;
      if (n < Nvalid) {
        long mb = m0 + wm * 32 + mt * 16 + hi * 8;
        #pragma unroll
        for (int r = 0; r < 8; ++r)
          C[(mb + r) * (long)ldc + n] = accs[mt][nt][r];
      }
    }
  }
}

// ---------------------------------------------------------------------------
// Weight f32 -> bf16 with zero row padding (dst [padRows][cols])
// ---------------------------------------------------------------------------
__global__ void cast_pad_kernel(unsigned short* __restrict__ dst,
                                const float* __restrict__ src,
                                int rows, int cols, int padRows)
{
  long idx = (long)blockIdx.x * 256 + threadIdx.x;
  long total = (long)padRows * cols;
  if (idx >= total) return;
  int n = (int)(idx / cols);
  dst[idx] = (n < rows) ? f2bf(src[idx]) : (unsigned short)0;
}

// ---------------------------------------------------------------------------
// cross_scan: x (4,192,64,64) -> xs bf16 [(b*4+k)*4096 + l][192]
// ---------------------------------------------------------------------------
__global__ void cross_scan_kernel(unsigned short* __restrict__ xs,
                                  const float* __restrict__ x)
{
  long idx = (long)blockIdx.x * 256 + threadIdx.x;   // 16*4096*192 exact
  int c = (int)(idx % 192);
  long t = idx / 192;
  int l  = (int)(t % 4096);
  int bk = (int)(t / 4096);
  int b = bk >> 2, k = bk & 3;
  int pos;
  if (k == 0)      pos = l;
  else if (k == 1) pos = ((l & 63) << 6) | (l >> 6);            // l = w*H+h -> h*W+w
  else if (k == 2) pos = 4095 - l;
  else { int l2 = 4095 - l; pos = ((l2 & 63) << 6) | (l2 >> 6); }
  xs[idx] = f2bf(x[((long)(b * 192 + c)) * 4096 + pos]);
}

// ---------------------------------------------------------------------------
// outer depthwise conv (SAME, k=3) + SiLU on x_ssm = proj[:, :384] -> bf16
// ---------------------------------------------------------------------------
__global__ void dwconv_silu_kernel(unsigned short* __restrict__ ubf,
                                   const float* __restrict__ proj,
                                   const float* __restrict__ w)
{
  long idx = (long)blockIdx.x * 256 + threadIdx.x;   // 65536*384 exact
  int c = (int)(idx % 384);
  long row = idx / 384;
  int l = (int)(row % 4096);
  const float* p = proj + row * 768 + c;
  float xm = (l > 0)    ? p[-768] : 0.f;
  float xc = p[0];
  float xp = (l < 4095) ? p[768]  : 0.f;
  float v = w[c * 3] * xm + w[c * 3 + 1] * xc + w[c * 3 + 2] * xp;
  ubf[idx] = f2bf(siluf(v));
}

// ---------------------------------------------------------------------------
// mamba causal depthwise conv (left pad 2, k=3) + bias + SiLU on xBC (416 ch)
// zx layout [row][896]: z=0..383, xBC=384..799, dt=800..805
// ---------------------------------------------------------------------------
__global__ void xbc_conv_kernel(float* __restrict__ xbcp,
                                const float* __restrict__ zx,
                                const float* __restrict__ w,
                                const float* __restrict__ bias)
{
  long idx = (long)blockIdx.x * 256 + threadIdx.x;   // 65536*416 exact
  int ch = (int)(idx % 416);
  long row = idx / 416;
  int l = (int)(row % 4096);
  float v = bias[ch];
  #pragma unroll
  for (int t = 0; t < 3; ++t) {
    int lp = l - 2 + t;
    if (lp >= 0) v += w[ch * 3 + t] * zx[(row + (long)(lp - l)) * 896 + 384 + ch];
  }
  xbcp[idx] = siluf(v);
}

// ---------------------------------------------------------------------------
// Selective scan: one block per (bk, head); state h[P=64][N=16] in registers
// (4 f32 per thread), y reduction over N via width-4 shfl_xor.
// ---------------------------------------------------------------------------
__global__ __launch_bounds__(256) void scan_kernel(
    float* __restrict__ yraw, const float* __restrict__ xbcp,
    const float* __restrict__ zx, const float* __restrict__ dt_bias,
    const float* __restrict__ A_log, const float* __restrict__ Dp)
{
  const int bk = blockIdx.x / 6;
  const int hh = blockIdx.x % 6;
  const int tid = threadIdx.x;
  const int p = tid >> 2;
  const int g = tid & 3;
  __shared__ float sx[64], sB[16], sC[16], sdt;
  float hs0 = 0.f, hs1 = 0.f, hs2 = 0.f, hs3 = 0.f;
  const float Ah = -__expf(A_log[hh]);
  const float dbias = dt_bias[hh];
  const float Dh = Dp[hh];
  const long base = (long)bk * 4096;

  for (int l = 0; l < 4096; ++l) {
    long row = base + l;
    const float* xr = xbcp + row * 416;
    if (tid < 64)        sx[tid]      = xr[hh * 64 + tid];
    else if (tid < 80)   sB[tid - 64] = xr[384 + (tid - 64)];
    else if (tid < 96)   sC[tid - 80] = xr[400 + (tid - 80)];
    else if (tid == 96)  sdt          = zx[row * 896 + 800 + hh];
    __syncthreads();

    float dtr = sdt + dbias;
    float dtv = (dtr > 20.f) ? dtr : log1pf(__expf(dtr));   // softplus
    float dA  = __expf(dtv * Ah);
    float xt  = sx[p];
    float coef = dtv * xt;
    int nb = g * 4;
    hs0 = hs0 * dA + coef * sB[nb];     float ys = hs0 * sC[nb];
    hs1 = hs1 * dA + coef * sB[nb + 1]; ys += hs1 * sC[nb + 1];
    hs2 = hs2 * dA + coef * sB[nb + 2]; ys += hs2 * sC[nb + 2];
    hs3 = hs3 * dA + coef * sB[nb + 3]; ys += hs3 * sC[nb + 3];
    ys += __shfl_xor(ys, 1, 32);
    ys += __shfl_xor(ys, 2, 32);
    if (g == 0) yraw[row * 384 + hh * 64 + p] = ys + xt * Dh;
    __syncthreads();
  }
}

// ---------------------------------------------------------------------------
// gated RMSNorm: y = y*silu(z); y *= rsqrt(mean(y^2)+1e-5)*norm_w -> bf16
// ---------------------------------------------------------------------------
__global__ __launch_bounds__(256) void gated_rmsnorm_kernel(
    unsigned short* __restrict__ ybf, const float* __restrict__ yraw,
    const float* __restrict__ zx, const float* __restrict__ nw)
{
  long row = blockIdx.x;
  int tid = threadIdx.x;
  __shared__ float red[256];
  const float* yr = yraw + row * 384;
  const float* zr = zx + row * 896;
  float g0 = yr[tid] * siluf(zr[tid]);
  float g1 = (tid < 128) ? yr[tid + 256] * siluf(zr[tid + 256]) : 0.f;
  red[tid] = g0 * g0 + g1 * g1;
  __syncthreads();
  for (int s = 128; s > 0; s >>= 1) {
    if (tid < s) red[tid] += red[tid + s];
    __syncthreads();
  }
  float rinv = rsqrtf(red[0] / 384.f + 1e-5f);
  ybf[row * 384 + tid] = f2bf(g0 * rinv * nw[tid]);
  if (tid < 128) ybf[row * 384 + tid + 256] = f2bf(g1 * rinv * nw[tid + 256]);
}

// ---------------------------------------------------------------------------
// gate with silu(x_gate) + cross_merge of the 4 scan directions -> bf16 rows
// ---------------------------------------------------------------------------
__global__ void merge_kernel(unsigned short* __restrict__ mbf,
                             const float* __restrict__ yproj,
                             const float* __restrict__ proj)
{
  long idx = (long)blockIdx.x * 256 + threadIdx.x;   // 16384*384 exact
  int c = (int)(idx % 384);
  long row = idx / 384;            // b*L + hw
  int hw = (int)(row % 4096);
  int b  = (int)(row / 4096);
  int h = hw >> 6, w0 = hw & 63;
  int l1 = (w0 << 6) | h;
  long r0 = (long)(b * 4 + 0) * 4096 + hw;
  long r2 = (long)(b * 4 + 2) * 4096 + (4095 - hw);
  long r1 = (long)(b * 4 + 1) * 4096 + l1;
  long r3 = (long)(b * 4 + 3) * 4096 + (4095 - l1);
  float v = yproj[r0 * 384 + c] * siluf(proj[r0 * 768 + 384 + c])
          + yproj[r2 * 384 + c] * siluf(proj[r2 * 768 + 384 + c])
          + yproj[r1 * 384 + c] * siluf(proj[r1 * 768 + 384 + c])
          + yproj[r3 * 384 + c] * siluf(proj[r3 * 768 + 384 + c]);
  mbf[idx] = f2bf(v);
}

// ---------------------------------------------------------------------------
// LayerNorm over C=192 + scaled residual; output (B,C,H,W) f32
// ---------------------------------------------------------------------------
__global__ __launch_bounds__(256) void ln_residual_kernel(
    float* __restrict__ out, const float* __restrict__ mproj,
    const float* __restrict__ x, const float* __restrict__ lw,
    const float* __restrict__ lb, const float* __restrict__ rs)
{
  long row = blockIdx.x;   // b*L + hw
  int tid = threadIdx.x;
  int hw = (int)(row % 4096);
  int b  = (int)(row / 4096);
  __shared__ float red[256];
  float m = (tid < 192) ? mproj[row * 256 + tid] : 0.f;
  red[tid] = m;
  __syncthreads();
  for (int s = 128; s > 0; s >>= 1) { if (tid < s) red[tid] += red[tid + s]; __syncthreads(); }
  float mu = red[0] / 192.f;
  __syncthreads();
  float d = (tid < 192) ? (m - mu) : 0.f;
  red[tid] = d * d;
  __syncthreads();
  for (int s = 128; s > 0; s >>= 1) { if (tid < s) red[tid] += red[tid + s]; __syncthreads(); }
  float var = red[0] / 192.f;
  if (tid < 192) {
    float val = (m - mu) * rsqrtf(var + 1e-5f) * lw[tid] + lb[tid];
    long xi = ((long)(b * 192 + tid)) * 4096 + hw;
    out[xi] = x[xi] + rs[0] * val;
  }
}

// ---------------------------------------------------------------------------
extern "C" void kernel_launch(void* const* d_in, const int* in_sizes, int n_in,
                              void* d_out, int out_size, void* d_ws, size_t ws_size,
                              hipStream_t stream)
{
  (void)in_sizes; (void)n_in; (void)out_size; (void)ws_size;
  const float* x           = (const float*)d_in[0];
  const float* in_proj_w   = (const float*)d_in[1];
  const float* conv1d_w    = (const float*)d_in[2];
  const float* m_in_proj_w = (const float*)d_in[3];
  const float* m_conv_w    = (const float*)d_in[4];
  const float* m_conv_b    = (const float*)d_in[5];
  const float* m_dt_bias   = (const float*)d_in[6];
  const float* m_A_log     = (const float*)d_in[7];
  const float* m_D         = (const float*)d_in[8];
  const float* m_norm_w    = (const float*)d_in[9];
  const float* m_out_proj_w= (const float*)d_in[10];
  const float* out_proj_w  = (const float*)d_in[11];
  const float* ln_w        = (const float*)d_in[12];
  const float* ln_b        = (const float*)d_in[13];
  const float* rscale      = (const float*)d_in[14];
  float* out = (float*)d_out;
  char* ws = (char*)d_ws;

  size_t off = 0;
  auto take = [&](size_t bytes) { size_t o = off; off += (bytes + 255) & ~(size_t)255; return o; };
  unsigned short* W1bf   = (unsigned short*)(ws + take((size_t)768 * 192 * 2));
  unsigned short* W2bf   = (unsigned short*)(ws + take((size_t)896 * 384 * 2));
  unsigned short* Wobf   = (unsigned short*)(ws + take((size_t)384 * 384 * 2));
  unsigned short* Woutbf = (unsigned short*)(ws + take((size_t)256 * 384 * 2));
  unsigned short* xsbf   = (unsigned short*)(ws + take((size_t)65536 * 192 * 2));
  float*          proj   = (float*)(ws + take((size_t)65536 * 768 * 4));
  unsigned short* ubf    = (unsigned short*)(ws + take((size_t)65536 * 384 * 2));
  float*          zx     = (float*)(ws + take((size_t)65536 * 896 * 4));
  float*          xbcp   = (float*)(ws + take((size_t)65536 * 416 * 4));
  float*          yraw   = (float*)(ws + take((size_t)65536 * 384 * 4));
  unsigned short* ybf    = (unsigned short*)(ws + take((size_t)65536 * 384 * 2));
  float*          yproj  = (float*)(ws + take((size_t)65536 * 384 * 4));
  unsigned short* mbf    = (unsigned short*)(ws + take((size_t)16384 * 384 * 2));
  float*          mproj  = (float*)(ws + take((size_t)16384 * 256 * 4));

  // weights -> bf16 (zero-padded N rows for WMMA tiling)
  cast_pad_kernel<<<(768 * 192 + 255) / 256, 256, 0, stream>>>(W1bf, in_proj_w, 768, 192, 768);
  cast_pad_kernel<<<(896 * 384 + 255) / 256, 256, 0, stream>>>(W2bf, m_in_proj_w, 806, 384, 896);
  cast_pad_kernel<<<(384 * 384 + 255) / 256, 256, 0, stream>>>(Wobf, m_out_proj_w, 384, 384, 384);
  cast_pad_kernel<<<(256 * 384 + 255) / 256, 256, 0, stream>>>(Woutbf, out_proj_w, 192, 384, 256);

  cross_scan_kernel<<<49152, 256, 0, stream>>>(xsbf, x);
  gemm_bf16_wmma<<<dim3(1024, 6), 256, 0, stream>>>(xsbf, W1bf, proj, 192, 768, 768);
  dwconv_silu_kernel<<<98304, 256, 0, stream>>>(ubf, proj, conv1d_w);
  gemm_bf16_wmma<<<dim3(1024, 7), 256, 0, stream>>>(ubf, W2bf, zx, 384, 896, 896);
  xbc_conv_kernel<<<106496, 256, 0, stream>>>(xbcp, zx, m_conv_w, m_conv_b);
  scan_kernel<<<96, 256, 0, stream>>>(yraw, xbcp, zx, m_dt_bias, m_A_log, m_D);
  gated_rmsnorm_kernel<<<65536, 256, 0, stream>>>(ybf, yraw, zx, m_norm_w);
  gemm_bf16_wmma<<<dim3(1024, 3), 256, 0, stream>>>(ybf, Wobf, yproj, 384, 384, 384);
  merge_kernel<<<24576, 256, 0, stream>>>(mbf, yproj, proj);
  gemm_bf16_wmma<<<dim3(256, 2), 256, 0, stream>>>(mbf, Woutbf, mproj, 384, 256, 256);
  ln_residual_kernel<<<16384, 256, 0, stream>>>(out, mproj, x, ln_w, ln_b, rscale);
}
